// HopfieldRepresentation_46926812676317
// MI455X (gfx1250) — matrix operand order repaired and checked by reading, not compiled, not run on hardware
//
#include <hip/hip_runtime.h>
#include <cstdint>
#include <cstddef>

// ---------------- CDNA5 types ----------------
typedef __attribute__((ext_vector_type(16))) __bf16 v16bf;
typedef __attribute__((ext_vector_type(8)))  __bf16 v8bf;
typedef __attribute__((ext_vector_type(8)))  float  v8f;
typedef __attribute__((ext_vector_type(4)))  float  v4f;

union V16 { v16bf v; v8bf h[2]; };

#define DIM   1024          // INPUT_DIM == HIDDEN_DIM
#define NH    16            // heads
#define NK    4096          // keys per head
#define NJ    (NK*NH)       // 65536 total key rows
#define NB    1024          // batch
// exp(beta * (x.k)/sqrt(D)) = exp2( acc * beta/sqrt(D) * log2(e) )
#define EXP2_SCALE 0.36067376022224085f

// ---------------- CDNA5 async global->LDS helpers ----------------
static __device__ __forceinline__ unsigned lds_off(const void* p) {
  // shared-space pointer: low 32 bits are the wave-relative LDS byte address
  return (unsigned)(uintptr_t)p;
}
static __device__ __forceinline__ void async_b128(const void* gptr, unsigned loff) {
  asm volatile("global_load_async_to_lds_b128 %0, %1, off"
               :: "v"(loff), "v"(gptr) : "memory");
}
static __device__ __forceinline__ void wait_async0() {
  asm volatile("s_wait_asynccnt 0" ::: "memory");
}

// ---------------- K0a: f32 -> bf16 elementwise (x, keys) ----------------
__global__ void k_cvt8(const float* __restrict__ in, __bf16* __restrict__ out, int n8) {
  int t = blockIdx.x * blockDim.x + threadIdx.x;
  if (t >= n8) return;
  const v4f* p = (const v4f*)(in + (size_t)t * 8);
  v4f a = p[0], b = p[1];
  v8bf o;
#pragma unroll
  for (int i = 0; i < 4; ++i) o[i] = (__bf16)a[i];
#pragma unroll
  for (int i = 0; i < 4; ++i) o[4 + i] = (__bf16)b[i];
  *(v8bf*)(out + (size_t)t * 8) = o;
}

// ---------------- K0b: values [NJ x DIM] f32 -> transposed bf16 [DIM x NJ] ----------------
__global__ void k_transcvt(const float* __restrict__ v, __bf16* __restrict__ vt) {
  __shared__ float tile[32][33];
  int jb = blockIdx.x * 32;              // key-row block
  int hb = blockIdx.y * 32;              // hidden-col block
  int lx = threadIdx.x & 31;
  int ly = threadIdx.x >> 5;             // 0..7
#pragma unroll
  for (int r = 0; r < 32; r += 8)
    tile[ly + r][lx] = v[(size_t)(jb + ly + r) * DIM + hb + lx];
  __syncthreads();
#pragma unroll
  for (int r = 0; r < 32; r += 8)
    vt[(size_t)(hb + ly + r) * NJ + jb + lx] = (__bf16)tile[lx][ly + r];
}

// ---------------- K1: GEMM1 + exp -> P (bf16), partial Z ----------------
// grid (16 rowblocks of 64 rows, 16 j-chunks of 4096 j), block 128 (4 waves)
// Each wave: one 16-row tile x 64 j-cols (4 WMMA accum tiles, A reused x4).
// LDS: double-buffered (A 64x128 + B 64x128) bf16 = 64KB -> ~5 WGs / WGP.
__global__ void k_phase1(const __bf16* __restrict__ xb, const __bf16* __restrict__ kb,
                         __bf16* __restrict__ P, float* __restrict__ Zp) {
  extern __shared__ char smem[];
  __bf16* As = (__bf16*)smem;            // 2 * 64*128 elems
  __bf16* Bs = As + 2 * 64 * 128;        // 2 * 64*128 elems

  const int rb    = blockIdx.x;
  const int chunk = blockIdx.y;
  const int tid   = threadIdx.x;
  const int wave  = tid >> 5;
  const int lane  = tid & 31;
  const int j0chunk = chunk * 4096;

  auto load_slab = [&](int step, int buf) {   // step = jblock*8 + kslab, 0..511
    int jblock = step >> 3, kslab = step & 7;
    const __bf16* ga = xb + (size_t)(rb * 64) * DIM + kslab * 128;
    const __bf16* gb = kb + (size_t)(j0chunk + jblock * 64) * DIM + kslab * 128;
    __bf16* ad = As + buf * 64 * 128;
    __bf16* bd = Bs + buf * 64 * 128;
#pragma unroll
    for (int i = 0; i < 8; ++i) {
      int c = tid * 8 + i;                    // 0..1023 16B chunks
      int row = c >> 4, seg = c & 15;
      async_b128(ga + (size_t)row * DIM + seg * 8, lds_off(ad + row * 128 + seg * 8));
      async_b128(gb + (size_t)row * DIM + seg * 8, lds_off(bd + row * 128 + seg * 8));
    }
  };

  const int aoff0 = (lane & 16) ? 16 : 0;     // A K-chunk select
  const int boff0 = (lane & 16) ? 16 : 0;     // B K-half select
  const int arow  = wave * 16 + (lane & 15);

  float zacc[8];
#pragma unroll
  for (int i = 0; i < 8; ++i) zacc[i] = 0.f;

  load_slab(0, 0);
  wait_async0();
  __syncthreads();

  for (int jblock = 0; jblock < 64; ++jblock) {
    v8f acc[4];
#pragma unroll
    for (int jt = 0; jt < 4; ++jt) acc[jt] = (v8f){};

    for (int kslab = 0; kslab < 8; ++kslab) {
      int step = jblock * 8 + kslab;
      int cur = step & 1;
      if (step + 1 < 512) load_slab(step + 1, cur ^ 1);
      const __bf16* a0 = As + cur * 64 * 128;
      const __bf16* b0 = Bs + cur * 64 * 128;
#pragma unroll
      for (int ks = 0; ks < 4; ++ks) {
        // hoist all operand loads ahead of the WMMA quartet so the
        // scheduler can pipeline ds_loads against v_wmma issue
        V16 a;
        {
          const __bf16* ap = a0 + (size_t)arow * 128 + ks * 32 + aoff0;
          a.h[0] = *(const v8bf*)ap;
          a.h[1] = *(const v8bf*)(ap + 8);
        }
        V16 b[4];
#pragma unroll
        for (int jt = 0; jt < 4; ++jt) {
          const __bf16* bp = b0 + (size_t)(jt * 16 + (lane & 15)) * 128
                                + ks * 32 + boff0;
          b[jt].h[0] = *(const v8bf*)bp;
          b[jt].h[1] = *(const v8bf*)(bp + 8);
        }
#pragma unroll
        for (int jt = 0; jt < 4; ++jt)
          acc[jt] = __builtin_amdgcn_wmma_f32_16x16x32_bf16(
              false, a.v, false, b[jt].v, (short)0, acc[jt], false, false);
      }
      wait_async0();
      __syncthreads();
    }

    // epilogue for this jblock: exp, Z accumulation, store unnormalized P
    int mr = rb * 64 + wave * 16 + ((lane & 16) ? 8 : 0);
#pragma unroll
    for (int jt = 0; jt < 4; ++jt) {
      int j = j0chunk + jblock * 64 + jt * 16 + (lane & 15);
#pragma unroll
      for (int s = 0; s < 8; ++s) {
        float p = __builtin_amdgcn_exp2f(acc[jt][s] * EXP2_SCALE);
        zacc[s] += p;
        P[(size_t)(mr + s) * NJ + j] = (__bf16)p;
      }
    }
  }

  // write partial Z: layout [chunk][row][head]
  {
    int mr = rb * 64 + wave * 16 + ((lane & 16) ? 8 : 0);
    int h  = lane & 15;
    float* zp = Zp + ((size_t)chunk * NB + mr) * NH + h;
#pragma unroll
    for (int s = 0; s < 8; ++s) zp[(size_t)s * NH] = zacc[s];
  }
}

// ---------------- K1b: reduce partial Z across 16 chunks, invert ----------------
__global__ void k_zred(const float* __restrict__ Zp, float* __restrict__ invZ) {
  int i = blockIdx.x * blockDim.x + threadIdx.x;     // 0..16383 = row*16+head
  float s = 0.f;
#pragma unroll
  for (int c = 0; c < 16; ++c) s += Zp[(size_t)c * (NB * NH) + i];
  invZ[i] = 1.0f / s;
}

// ---------------- K2: P *= invZ[row, j%16] (in place, 8 bf16 per thread) ----------------
__global__ void k_scale(__bf16* __restrict__ P, const float* __restrict__ invZ) {
  size_t t = (size_t)blockIdx.x * blockDim.x + threadIdx.x;
  size_t e = t * 8;
  int m = (int)(e >> 16);
  int j = (int)(e & (NJ - 1));
  const float* z = invZ + (size_t)m * NH + (j & 8);  // heads j%16 .. +7
  v8bf* p = (v8bf*)(P + e);
  v8bf a = *p, o;
#pragma unroll
  for (int i = 0; i < 8; ++i) o[i] = (__bf16)((float)a[i] * z[i]);
  *p = o;
}

// ---------------- K3: out = Pn @ Vt^T, 128x128 tiles, K-slabs of 128 ----------------
// grid (8,8), block 256 (8 waves: 4 row-groups x 2 col-groups)
// LDS: A 2x(128x128) bf16 + B 2x(128x128) bf16 = 128KB -> 2 WGs / WGP.
__global__ void k_gemm2(const __bf16* __restrict__ P, const __bf16* __restrict__ Vt,
                        float* __restrict__ out) {
  extern __shared__ char smem[];
  __bf16* As = (__bf16*)smem;                        // 2 * 16384 elems
  __bf16* Bs = As + 2 * 128 * 128;

  const int rb = blockIdx.x, cb = blockIdx.y;
  const int tid  = threadIdx.x;
  const int wave = tid >> 5, lane = tid & 31;
  const int rg = wave >> 1, cg = wave & 1;

  v8f acc[2][4];
#pragma unroll
  for (int i = 0; i < 2; ++i)
#pragma unroll
    for (int k = 0; k < 4; ++k) acc[i][k] = (v8f){};

  auto load_slab = [&](int slab, int buf) {
    size_t j0 = (size_t)slab * 128;
    __bf16* ad = As + buf * 128 * 128;
    __bf16* bd = Bs + buf * 128 * 128;
#pragma unroll
    for (int i = 0; i < 8; ++i) {
      int c = tid * 8 + i;                           // 0..2047
      int row = c >> 4, seg = c & 15;
      async_b128(P  + (size_t)(rb * 128 + row) * NJ + j0 + seg * 8,
                 lds_off(ad + row * 128 + seg * 8));
      async_b128(Vt + (size_t)(cb * 128 + row) * NJ + j0 + seg * 8,
                 lds_off(bd + row * 128 + seg * 8));
    }
  };

  load_slab(0, 0);
  wait_async0();
  __syncthreads();

  const int aoff0 = (lane & 16) ? 16 : 0;
  const int boff0 = (lane & 16) ? 16 : 0;

  for (int slab = 0; slab < NJ / 128; ++slab) {
    int cur = slab & 1;
    if (slab + 1 < NJ / 128) load_slab(slab + 1, cur ^ 1);
    const __bf16* a0 = As + cur * 128 * 128;
    const __bf16* b0 = Bs + cur * 128 * 128;
#pragma unroll
    for (int ks = 0; ks < 4; ++ks) {
      V16 a[2], b[4];
#pragma unroll
      for (int rt = 0; rt < 2; ++rt) {
        const __bf16* ap = a0 + (size_t)(rg * 32 + rt * 16 + (lane & 15)) * 128
                              + ks * 32 + aoff0;
        a[rt].h[0] = *(const v8bf*)ap;
        a[rt].h[1] = *(const v8bf*)(ap + 8);
      }
#pragma unroll
      for (int ct = 0; ct < 4; ++ct) {
        const __bf16* bp = b0 + (size_t)(cg * 64 + ct * 16 + (lane & 15)) * 128
                              + ks * 32 + boff0;
        b[ct].h[0] = *(const v8bf*)bp;
        b[ct].h[1] = *(const v8bf*)(bp + 8);
      }
#pragma unroll
      for (int rt = 0; rt < 2; ++rt)
#pragma unroll
        for (int ct = 0; ct < 4; ++ct)
          acc[rt][ct] = __builtin_amdgcn_wmma_f32_16x16x32_bf16(
              false, a[rt].v, false, b[ct].v, (short)0, acc[rt][ct], false, false);
    }
    wait_async0();
    __syncthreads();
  }

  // store f32 output
#pragma unroll
  for (int rt = 0; rt < 2; ++rt) {
    int r0 = rb * 128 + rg * 32 + rt * 16 + ((lane & 16) ? 8 : 0);
#pragma unroll
    for (int ct = 0; ct < 4; ++ct) {
      int c0 = cb * 128 + cg * 64 + ct * 16 + (lane & 15);
#pragma unroll
      for (int s = 0; s < 8; ++s)
        out[(size_t)(r0 + s) * DIM + c0] = acc[rt][ct][s];
    }
  }
}

// ---------------- host launcher ----------------
extern "C" void kernel_launch(void* const* d_in, const int* in_sizes, int n_in,
                              void* d_out, int out_size, void* d_ws, size_t ws_size,
                              hipStream_t stream) {
  (void)in_sizes; (void)n_in; (void)out_size; (void)ws_size;
  const float* x      = (const float*)d_in[0];
  const float* keys   = (const float*)d_in[1];
  const float* values = (const float*)d_in[2];
  float* out = (float*)d_out;

  char* ws = (char*)d_ws;
  const size_t MB = 1u << 20;
  __bf16* xb   = (__bf16*)(ws);                        // 2 MB
  __bf16* kb   = (__bf16*)(ws + 2 * MB);               // 128 MB
  __bf16* vt   = (__bf16*)(ws + 130 * MB);             // 128 MB (transposed values)
  __bf16* P    = (__bf16*)(ws + 258 * MB);             // 128 MB
  float*  Zp   = (float*) (ws + 386 * MB);             // 1 MB
  float*  invZ = (float*) (ws + 387 * MB);             // 64 KB

  // K0: conversions
  k_cvt8<<<(NB * DIM / 8) / 256, 256, 0, stream>>>(x, xb, NB * DIM / 8);
  k_cvt8<<<((size_t)NJ * DIM / 8) / 256, 256, 0, stream>>>(keys, kb, NJ * DIM / 8);
  k_transcvt<<<dim3(NJ / 32, DIM / 32), 256, 0, stream>>>(values, vt);

  // K1: GEMM1 + exp -> P, partial Z   (LDS: 64KB double-buffered slabs)
  k_phase1<<<dim3(16, 16), 128, 65536, stream>>>(xb, kb, P, Zp);

  // K1b: reduce Z, invert
  k_zred<<<(NB * NH) / 256, 256, 0, stream>>>(Zp, invZ);

  // K2: normalize P in place
  k_scale<<<(int)(((size_t)NB * NJ / 8) / 256), 256, 0, stream>>>(P, invZ);

  // K3: out = Pn @ Vt^T   (LDS: 128KB)
  k_gemm2<<<dim3(NB / 128, DIM / 128), 256, 0, stream>>>(P, vt, out);
}
